// TorchFastSuperModel_89386859364560
// MI455X (gfx1250) — matrix operand order repaired and checked by reading, not compiled, not run on hardware
//
#include <hip/hip_runtime.h>

typedef __attribute__((ext_vector_type(16))) _Float16 v16h;
typedef __attribute__((ext_vector_type(8)))  float    v8f;

#define CDIM 32
#define KTOT 16
#define ZDIM 64
#define NVOX (2*96*96*64)          // 1179648 voxels
#define NTILES (NVOX/16)           // 73728 wave tiles
#define NBLOCKS (NTILES/8)         // 9216 blocks of 8 waves
#define NCHUNK 33                  // 32 quadratic chunks + 1 linear chunk
#define SCLF (-0.5f / (200.0f * 0.69314718055994531f))  // logp/200 -> exp2 arg

static __device__ __forceinline__ v16h splat16(_Float16 s) {
    v16h v;
    #pragma unroll
    for (int i = 0; i < 16; ++i) v[i] = s;
    return v;
}

// ---------------------------------------------------------------------------
// Prep: invert the 16 Cholesky factors, then emit B = [SCL*P_k | SCL*q_k]
// (f16, pre-swizzled into the WMMA B lane/VGPR layout, components on the
// N axis) and coef[k] = w_k * exp((const - logdet_k)/200) * 2^(SCL*r_k),
// where P = Linv^T Linv, q = -2 P mu, r = ||Linv mu||^2.
// ---------------------------------------------------------------------------
__global__ __launch_bounds__(512) void gmm_prep_kernel(
    const float* __restrict__ wbg, const float* __restrict__ mbg,
    const float* __restrict__ lbg, const float* __restrict__ wfg,
    const float* __restrict__ mfg, const float* __restrict__ lfg,
    float* __restrict__ ws_coef, _Float16* __restrict__ ws_b)
{
    __shared__ float sLinv[KTOT][CDIM][CDIM];   // 64 KB
    const int tid = threadIdx.x;
    const int k = tid >> 5;        // component 0..15
    const int j = tid & 31;        // column of L^{-1}

    const float* Lk = (k < 8) ? (lbg + k * CDIM * CDIM)
                              : (lfg + (k - 8) * CDIM * CDIM);
    for (int i = 0; i < j; ++i) sLinv[k][i][j] = 0.0f;   // strict upper = 0
    for (int i = j; i < CDIM; ++i) {                     // forward subst.
        float s = (i == j) ? 1.0f : 0.0f;
        for (int p = j; p < i; ++p) s -= Lk[i * CDIM + p] * sLinv[k][p][j];
        sLinv[k][i][j] = s / Lk[i * CDIM + i];
    }
    __syncthreads();

    // coef[k]: fold weight, normalizer, logdet and constant term r
    if (tid < KTOT) {
        const float* Lc = (tid < 8) ? (lbg + tid * CDIM * CDIM)
                                    : (lfg + (tid - 8) * CDIM * CDIM);
        const float* mu = (tid < 8) ? (mbg + tid * CDIM) : (mfg + (tid - 8) * CDIM);
        float ld = 0.0f;
        for (int i = 0; i < CDIM; ++i) ld += logf(Lc[i * CDIM + i]);
        float r = 0.0f;
        for (int d = 0; d < CDIM; ++d) {
            float vd = 0.0f;
            for (int c = 0; c < CDIM; ++c) vd += sLinv[tid][d][c] * mu[c];
            r += vd * vd;
        }
        const float CONSTT = -16.0f * 1.8378770664093453f; // -0.5*C*log(2pi)
        float w = (tid < 8) ? wbg[tid] : wfg[tid - 8];
        ws_coef[tid] = w * expf((CONSTT - ld) / 200.0f) * exp2f(SCLF * r);
    }

    // B tiles: chunk t<32 -> SCL*P_n[t][c'] ; chunk 32 -> SCL*q_n[c'].
    // Entry e = t*512 + l*16 + i : lane l, column n = l&15 (component),
    // K-in-chunk c' = (l&16 ? 8 : 0) + (i < 8 ? i : i + 8).
    for (int e = tid; e < NCHUNK * 512; e += 512) {
        const int t = e >> 9;
        const int rem = e & 511;
        const int l = rem >> 4;
        const int i = rem & 15;
        const int n = l & 15;
        const int cp = ((l & 16) ? 8 : 0) + ((i < 8) ? i : (i + 8));
        float val;
        if (t < 32) {
            float s = 0.0f;
            for (int d = 0; d < CDIM; ++d)
                s += sLinv[n][d][t] * sLinv[n][d][cp];
            val = SCLF * s;                      // SCL * P[t][c']
        } else {
            const float* mu = (n < 8) ? (mbg + n * CDIM) : (mfg + (n - 8) * CDIM);
            float s = 0.0f;
            for (int d = 0; d < CDIM; ++d) {
                float vd = 0.0f;
                for (int c = 0; c < CDIM; ++c) vd += sLinv[n][d][c] * mu[c];
                s += sLinv[n][d][cp] * vd;
            }
            val = SCLF * (-2.0f) * s;            // SCL * q[c']
        }
        ws_b[e] = (_Float16)val;
    }
}

// ---------------------------------------------------------------------------
// Main: wave32 owns 16 voxels. A chunks = x_t * (f16 feature vector)
// (outer-product terms, 8 v_pk_mul_f16 each) + one linear chunk; 33 chained
// WMMAs accumulate D[voxel][component] = SCL*(x P x + q x) directly.
// Epilogue: 8 v_exp + ONE packed butterfly reduction per wave-tile.
// ---------------------------------------------------------------------------
__global__ __launch_bounds__(256) void gmm_density_kernel(
    const float* __restrict__ feat, const float* __restrict__ ws_coef,
    const _Float16* __restrict__ ws_b, float* __restrict__ out)
{
    __shared__ __align__(32) _Float16 sB[NCHUNK * 512];  // 33 KB of B tiles
    __shared__ float sCoef[KTOT];

    const int tid = threadIdx.x;
    #pragma unroll
    for (int i = tid; i < (NCHUNK * 512) / 8; i += 256)   // 2112 uint4
        ((uint4*)sB)[i] = ((const uint4*)ws_b)[i];
    if (tid < KTOT) sCoef[tid] = ws_coef[tid];
    __syncthreads();

    const int wave = tid >> 5;
    const int lane = tid & 31;
    const int tile = blockIdx.x * 8 + wave;      // grid exactly covers NTILES
    const int vrow = lane & 15;
    const int voxel = tile * 16 + vrow;
    const bool hih = (lane & 16) != 0;
    const float coefLane = sCoef[lane & 15];     // N column = component

    // all 32 features of this lane's voxel (lane pairs share cachelines)
    const float* fp = feat + (size_t)voxel * CDIM;
    float f[CDIM];
    #pragma unroll
    for (int q = 0; q < 8; ++q) {
        float4 v = ((const float4*)fp)[q];
        f[4*q] = v.x; f[4*q+1] = v.y; f[4*q+2] = v.z; f[4*q+3] = v.w;
    }

    // A-operand feature vector: low half K={0..7,16..23}, high {8..15,24..31}
    v16h af;
    #pragma unroll
    for (int i = 0; i < 16; ++i) {
        const int base = (i < 8) ? i : (i + 8);
        af[i] = (_Float16)(hih ? f[base + 8] : f[base]);
    }

    // 33 chained WMMAs: D accumulates SCL*(x^T P x + q^T x) per (voxel,comp)
    v8f acc = {};
    #pragma unroll
    for (int t = 0; t < 32; ++t) {
        v16h b = *(const v16h*)(&sB[t * 512 + lane * 16]);
        v16h av = af * splat16((_Float16)f[t]);          // 8 v_pk_mul_f16
        acc = __builtin_amdgcn_wmma_f32_16x16x32_f16(
            false, av, false, b, (short)0, acc, false, false);
    }
    {
        v16h b = *(const v16h*)(&sB[32 * 512 + lane * 16]);
        acc = __builtin_amdgcn_wmma_f32_16x16x32_f16(
            false, af, false, b, (short)0, acc, false, false);
    }

    // per-element contribution: coef_n * 2^acc  (8 v_exp total)
    float e0 = coefLane * __builtin_amdgcn_exp2f(acc[0]);
    float e1 = coefLane * __builtin_amdgcn_exp2f(acc[1]);
    float e2 = coefLane * __builtin_amdgcn_exp2f(acc[2]);
    float e3 = coefLane * __builtin_amdgcn_exp2f(acc[3]);
    float e4 = coefLane * __builtin_amdgcn_exp2f(acc[4]);
    float e5 = coefLane * __builtin_amdgcn_exp2f(acc[5]);
    float e6 = coefLane * __builtin_amdgcn_exp2f(acc[6]);
    float e7 = coefLane * __builtin_amdgcn_exp2f(acc[7]);

    // packed butterfly: sum over the 16 component-lanes, packing two source
    // registers per shuffle round (8 shuffles total instead of 32).
    float g0, g1, g2, g3, h0, h1, u;
    {   // round mask=8: 8 regs -> 4
        float x, y;
        x = (lane & 8) ? e4 : e0; y = (lane & 8) ? e0 : e4;
        g0 = x + __shfl_xor(y, 8, 32);
        x = (lane & 8) ? e5 : e1; y = (lane & 8) ? e1 : e5;
        g1 = x + __shfl_xor(y, 8, 32);
        x = (lane & 8) ? e6 : e2; y = (lane & 8) ? e2 : e6;
        g2 = x + __shfl_xor(y, 8, 32);
        x = (lane & 8) ? e7 : e3; y = (lane & 8) ? e3 : e7;
        g3 = x + __shfl_xor(y, 8, 32);
    }
    {   // round mask=4: 4 -> 2
        float x, y;
        x = (lane & 4) ? g2 : g0; y = (lane & 4) ? g0 : g2;
        h0 = x + __shfl_xor(y, 4, 32);
        x = (lane & 4) ? g3 : g1; y = (lane & 4) ? g1 : g3;
        h1 = x + __shfl_xor(y, 4, 32);
    }
    {   // round mask=2: 2 -> 1
        float x = (lane & 2) ? h1 : h0;
        float y = (lane & 2) ? h0 : h1;
        u = x + __shfl_xor(y, 2, 32);
    }
    u += __shfl_xor(u, 1, 32);   // final 16-lane sum

    // lane now holds dens(row), row = ((lane>>1)&7) + (hi ? 8 : 0);
    // each row duplicated across bit0 -> store from even lanes only.
    const int row = ((lane >> 1) & 7) + (hih ? 8 : 0);
    const int vox = tile * 16 + row;
    if (!(lane & 1)) {
        out[vox] = ((vox & (ZDIM - 1)) == ZDIM - 1) ? 0.0f : u;
    }
}

extern "C" void kernel_launch(void* const* d_in, const int* in_sizes, int n_in,
                              void* d_out, int out_size, void* d_ws, size_t ws_size,
                              hipStream_t stream) {
    const float* feat = (const float*)d_in[0];
    const float* wbg  = (const float*)d_in[1];
    const float* mbg  = (const float*)d_in[2];
    const float* lbg  = (const float*)d_in[3];
    const float* wfg  = (const float*)d_in[4];
    const float* mfg  = (const float*)d_in[5];
    const float* lfg  = (const float*)d_in[6];

    float*    ws_coef = (float*)d_ws;                      // 16 floats
    _Float16* ws_b    = (_Float16*)((char*)d_ws + 64);     // 16896 halves

    gmm_prep_kernel<<<1, 512, 0, stream>>>(wbg, mbg, lbg, wfg, mfg, lfg,
                                           ws_coef, ws_b);
    gmm_density_kernel<<<NBLOCKS, 256, 0, stream>>>(feat, ws_coef, ws_b,
                                                    (float*)d_out);
}
